// PointEncoder2_28656021799039
// MI455X (gfx1250) — matrix operand order repaired and checked by reading, not compiled, not run on hardware
//
#include <hip/hip_runtime.h>

typedef __attribute__((ext_vector_type(2)))  float   v2f;
typedef __attribute__((ext_vector_type(8)))  float   v8f;
typedef __attribute__((ext_vector_type(16))) __bf16  v16bf;

#define N_PTS  65536
#define PPB    16384
#define KSEL   4096
#define HID    256

__device__ __forceinline__ unsigned short f2bf(float f) {
  union { float f; unsigned u; } c; c.f = f;
  unsigned u = c.u;
  return (unsigned short)((u + 0x7FFFu + ((u >> 16) & 1u)) >> 16);
}
__device__ __forceinline__ float bf2f(unsigned short h) {
  union { unsigned u; float f; } c; c.u = ((unsigned)h) << 16;
  return c.f;
}
__device__ __forceinline__ unsigned fbits(float f) { union { float f; unsigned u; } c; c.f = f; return c.u; }
__device__ __forceinline__ float bitsf(unsigned u) { union { unsigned u; float f; } c; c.u = u; return c.f; }

union BFrag { v16bf v; unsigned short s[16]; uint4 q[2]; };

// ---------------- zero the BN accumulators (replay-safe) ----------------
__global__ void k_zero(float* g_acc) {
  if (threadIdx.x < 2) g_acc[threadIdx.x] = 0.f;
}

// ------- pre-shuffle w2 into f32 WMMA B-fragment layout (16x16x4) -------
// B 4x16 f32: lane -> N = lane&15 ; element e -> K = 4*kc + 2*(lane>>4) + e
__global__ __launch_bounds__(256) void k_prep_w2(const float* __restrict__ w2,
                                                 float* __restrict__ w2ff) {
  const int t = (int)blockIdx.x * 256 + threadIdx.x;     // 32768 threads
  const int nt = t >> 11, kc = (t >> 5) & 63, l = t & 31;
  const int n = nt * 16 + (l & 15);
  const int K = kc * 4 + ((l >> 4) << 1);
  float2 v;
  v.x = w2[K * HID + n];
  v.y = w2[(K + 1) * HID + n];
  ((float2*)w2ff)[(nt * 64 + kc) * 32 + l] = v;
}

// ------ pre-shuffle wp1 into bf16 WMMA B-fragment layout (16x16x32) -----
// B 32x16 bf16: lane -> N = lane&15 ; pair p -> K = 32*kc + 16*(lane>>4) + 2p + e
__global__ __launch_bounds__(256) void k_prep_wp1(const float* __restrict__ wp1,
                                                  unsigned short* __restrict__ wp1f) {
  const int t = (int)blockIdx.x * 256 + threadIdx.x;     // 4096 threads
  const int nt = t >> 8, kc = (t >> 5) & 7, l = t & 31;
  const int n = nt * 16 + (l & 15);
  const int kb = kc * 32 + ((l >> 4) << 4);
  BFrag f;
#pragma unroll
  for (int p = 0; p < 8; ++p) {
    const int K = kb + 2 * p;
    f.s[2 * p]     = f2bf(wp1[K * HID + n]);
    f.s[2 * p + 1] = f2bf(wp1[(K + 1) * HID + n]);
  }
  uint4* d = (uint4*)(wp1f + ((nt * 8 + kc) * 32 + l) * 16);
  d[0] = f.q[0];
  d[1] = f.q[1];
}

// --- fused: h=relu(x@w1+b1); pf=relu(h@w2+b2) via f32 WMMA; s=pf.ws+bs ---
__global__ __launch_bounds__(256) void k_mlp(
    const float* __restrict__ x,  const float* __restrict__ w1,
    const float* __restrict__ b1, const float* __restrict__ b2,
    const float* __restrict__ wsmp, const float* __restrict__ bsmp,
    const float* __restrict__ w2ff,
    unsigned short* __restrict__ pf_bf, float* __restrict__ s_buf,
    float* __restrict__ g_acc)
{
  const int tid = threadIdx.x;
  const int wv = tid >> 5, l = tid & 31, lo = l & 15, hi = l >> 4;
  const int m0 = (int)blockIdx.x * 128 + wv * 16;

  const float4 xv = ((const float4*)x)[m0 + lo];

  // hl[2*kc+e] == A-fragment element for (M=lo, K=4*kc+2*hi+e), kept f32.
  float hl[128];
#pragma unroll
  for (int kc = 0; kc < 64; ++kc) {
    const int K = kc * 4 + (hi << 1);
    const float2 bb = *(const float2*)(b1 + K);
    const float2 wa = *(const float2*)(w1 + K);
    const float2 wb = *(const float2*)(w1 + 256 + K);
    const float2 wc = *(const float2*)(w1 + 512 + K);
    const float2 wd = *(const float2*)(w1 + 768 + K);
    float h0 = bb.x + xv.x * wa.x + xv.y * wb.x + xv.z * wc.x + xv.w * wd.x;
    float h1 = bb.y + xv.x * wa.y + xv.y * wb.y + xv.z * wc.y + xv.w * wd.y;
    hl[2 * kc]     = h0 > 0.f ? h0 : 0.f;
    hl[2 * kc + 1] = h1 > 0.f ? h1 : 0.f;
  }

  float sacc[8];
#pragma unroll
  for (int r = 0; r < 8; ++r) sacc[r] = 0.f;

  for (int nt = 0; nt < 16; ++nt) {
    v8f acc;
#pragma unroll
    for (int r = 0; r < 8; ++r) acc[r] = 0.f;
    const float2* bsrc = ((const float2*)w2ff) + nt * 2048 + l;
#pragma unroll
    for (int kc = 0; kc < 64; ++kc) {
      const float2 bl = bsrc[kc * 32];
      v2f av; av[0] = hl[2 * kc]; av[1] = hl[2 * kc + 1];
      v2f bv; bv[0] = bl.x;       bv[1] = bl.y;
      acc = __builtin_amdgcn_wmma_f32_16x16x4_f32(false, av, false, bv,
                                                  (short)0, acc, false, false);
    }
    const int n = nt * 16 + lo;
    const float bias = b2[n];
    const float wsn  = wsmp[n];
#pragma unroll
    for (int r = 0; r < 8; ++r) {
      float v = acc[r] + bias;
      v = v > 0.f ? v : 0.f;
      const int row = m0 + r + (hi << 3);
      pf_bf[row * HID + n] = f2bf(v);
      sacc[r] += v * wsn;   // partial of s over this lane's n column
    }
  }

#pragma unroll
  for (int r = 0; r < 8; ++r) {
    sacc[r] += __shfl_xor(sacc[r], 1, 32);
    sacc[r] += __shfl_xor(sacc[r], 2, 32);
    sacc[r] += __shfl_xor(sacc[r], 4, 32);
    sacc[r] += __shfl_xor(sacc[r], 8, 32);
  }
  if (lo == 0) {
    const float bsv = bsmp[0];
    float ls = 0.f, lq = 0.f;
#pragma unroll
    for (int r = 0; r < 8; ++r) {
      const float sv = sacc[r] + bsv;
      s_buf[m0 + r + (hi << 3)] = sv;
      ls += sv; lq += sv * sv;
    }
    atomicAdd(&g_acc[0], ls);
    atomicAdd(&g_acc[1], lq);
  }
}

// ---------------- BN stats finalize ----------------
__global__ void k_bnstats(const float* __restrict__ g_acc, float* __restrict__ g_musd) {
  const float inv = 1.f / (float)N_PTS;
  const float mu = g_acc[0] * inv;
  const float var = g_acc[1] * inv - mu * mu;
  g_musd[0] = mu;
  g_musd[1] = rsqrtf(var + 1e-5f);
}

// ---------------- scores = relu(BN(s)) ----------------
__global__ __launch_bounds__(256) void k_score(const float* __restrict__ s_buf,
                                               const float* __restrict__ g_musd,
                                               const float* __restrict__ gamma,
                                               const float* __restrict__ beta,
                                               float* __restrict__ score) {
  const int i = (int)blockIdx.x * 256 + threadIdx.x;
  float sc = (s_buf[i] - g_musd[0]) * g_musd[1] * gamma[0] + beta[0];
  score[i] = sc > 0.f ? sc : 0.f;
}

// --- per-batch bitonic sort of 16384 (score desc, idx asc), emit top-4096
//     ranks + per-batch max score and sum of selected exp((s-max)/0.1) ---
extern __shared__ unsigned long long smem_sort[];
__global__ __launch_bounds__(1024) void k_sort(const float* __restrict__ score,
                                               int* __restrict__ topk,
                                               float* __restrict__ bmax,
                                               float* __restrict__ bsum) {
  unsigned long long* keys = smem_sort;
  float* red = (float*)(keys + PPB);
  const int t = threadIdx.x;
  const int b = blockIdx.x;

  for (int i = t; i < PPB; i += 1024) {
    const unsigned kb = fbits(score[b * PPB + i]);   // scores >= 0 -> bit order == value order
    keys[i] = ((unsigned long long)kb << 32) |
              (unsigned long long)(0xFFFFFFFFu - (unsigned)i); // ties: lower idx ranks higher
  }
  __syncthreads();

  for (unsigned k = 2; k <= PPB; k <<= 1) {
    for (unsigned j = k >> 1; j > 0; j >>= 1) {
      for (unsigned i = t; i < PPB; i += 1024) {
        const unsigned ixj = i ^ j;
        if (ixj > i) {
          const bool up = ((i & k) == 0);
          unsigned long long A = keys[i], B = keys[ixj];
          if ((A > B) == up) { keys[i] = B; keys[ixj] = A; }
        }
      }
      __syncthreads();
    }
  }

  const float mx = bitsf((unsigned)(keys[PPB - 1] >> 32));
  float lsum = 0.f;
  for (int i = t; i < KSEL; i += 1024) {
    const unsigned long long e = keys[PPB - 1 - i];
    const float sc = bitsf((unsigned)(e >> 32));
    const unsigned idx = 0xFFFFFFFFu - (unsigned)e;
    topk[b * KSEL + i] = b * PPB + (int)idx;           // global point index, rank order
    lsum += __expf((sc - mx) * 10.f);
  }
  red[t] = lsum;
  __syncthreads();
  for (int s = 512; s > 0; s >>= 1) {
    if (t < s) red[t] += red[t + s];
    __syncthreads();
  }
  if (t == 0) { bmax[b] = mx; bsum[b] = red[0]; }
}

// --- processor: out = relu((pf[sel]*sw) @ wp1 + bp1) @ wp2 + bp2
//     gathered A scaled+converted to bf16 fragments, bf16 WMMA ---
__global__ __launch_bounds__(256) void k_proc(
    const unsigned short* __restrict__ pf_bf, const int* __restrict__ topk,
    const float* __restrict__ score, const float* __restrict__ bmax,
    const float* __restrict__ bsum, const unsigned short* __restrict__ wp1f,
    const float* __restrict__ bp1, const float* __restrict__ wp2,
    const float* __restrict__ bp2, float* __restrict__ out)
{
  const int tid = threadIdx.x;
  const int wv = tid >> 5, l = tid & 31, lo = l & 15, hi = l >> 4;
  const int base = (int)blockIdx.x * 128 + wv * 16;
  const int gRow = base + lo;                 // selected-row index (b*4096 + rank)
  const int b = gRow >> 12;
  const int src = topk[gRow];
  const float sw = __expf((score[src] - bmax[b]) * 10.f) / bsum[b];
  const unsigned short* prow = pf_bf + src * HID;

  BFrag a[8];   // bf16 A fragments, 16x32 layout: K = 32*kc + 16*hi + 2p + e
#pragma unroll
  for (int kc = 0; kc < 8; ++kc) {
#pragma unroll
    for (int p = 0; p < 8; ++p) {
      const int K = kc * 32 + (p << 1) + (hi << 4);
      const unsigned u = *(const unsigned*)(prow + K);
      a[kc].s[2 * p]     = f2bf(bf2f((unsigned short)(u & 0xFFFFu)) * sw);
      a[kc].s[2 * p + 1] = f2bf(bf2f((unsigned short)(u >> 16)) * sw);
    }
  }

  float o[8][4];
#pragma unroll
  for (int r = 0; r < 8; ++r)
#pragma unroll
    for (int c = 0; c < 4; ++c) o[r][c] = 0.f;

  for (int nt = 0; nt < 16; ++nt) {
    v8f acc;
#pragma unroll
    for (int r = 0; r < 8; ++r) acc[r] = 0.f;
#pragma unroll
    for (int kc = 0; kc < 8; ++kc) {
      BFrag bf;
      const uint4* bp = (const uint4*)(wp1f + ((nt * 8 + kc) * 32 + l) * 16);
      bf.q[0] = bp[0];
      bf.q[1] = bp[1];
      acc = __builtin_amdgcn_wmma_f32_16x16x32_bf16(false, a[kc].v, false, bf.v,
                                                    (short)0, acc, false, false);
    }
    const int n = nt * 16 + lo;
    const float bias = bp1[n];
    const float4 wq = ((const float4*)wp2)[n];   // wp2 row n: 4 outputs
#pragma unroll
    for (int r = 0; r < 8; ++r) {
      float v = acc[r] + bias;
      v = v > 0.f ? v : 0.f;
      o[r][0] += v * wq.x; o[r][1] += v * wq.y;
      o[r][2] += v * wq.z; o[r][3] += v * wq.w;
    }
  }

#pragma unroll
  for (int r = 0; r < 8; ++r)
#pragma unroll
    for (int c = 0; c < 4; ++c) {
      o[r][c] += __shfl_xor(o[r][c], 1, 32);
      o[r][c] += __shfl_xor(o[r][c], 2, 32);
      o[r][c] += __shfl_xor(o[r][c], 4, 32);
      o[r][c] += __shfl_xor(o[r][c], 8, 32);
    }

  if (lo == 0) {
    const float4 bq = *(const float4*)bp2;
#pragma unroll
    for (int r = 0; r < 8; ++r) {
      const int row = base + r + (hi << 3);
      float4 ov;
      ov.x = o[r][0] + bq.x; ov.y = o[r][1] + bq.y;
      ov.z = o[r][2] + bq.z; ov.w = o[r][3] + bq.w;
      ((float4*)out)[row] = ov;
    }
  }
}

extern "C" void kernel_launch(void* const* d_in, const int* in_sizes, int n_in,
                              void* d_out, int out_size, void* d_ws, size_t ws_size,
                              hipStream_t stream) {
  (void)in_sizes; (void)n_in; (void)out_size; (void)ws_size;
  const float* x     = (const float*)d_in[0];
  const float* w1    = (const float*)d_in[1];
  const float* b1    = (const float*)d_in[2];
  const float* w2    = (const float*)d_in[3];
  const float* b2    = (const float*)d_in[4];
  const float* wsmp  = (const float*)d_in[5];
  const float* bsmp  = (const float*)d_in[6];
  const float* gamma = (const float*)d_in[7];
  const float* beta  = (const float*)d_in[8];
  const float* wp1   = (const float*)d_in[9];
  const float* bp1   = (const float*)d_in[10];
  const float* wp2   = (const float*)d_in[11];
  const float* bp2   = (const float*)d_in[12];
  float* out = (float*)d_out;

  char* ws = (char*)d_ws;
  float* g_acc  = (float*)(ws + 0);        // [2] sum, sumsq
  float* g_musd = (float*)(ws + 64);       // [2] mu, rstd
  float* g_bmax = (float*)(ws + 128);      // [4]
  float* g_bsum = (float*)(ws + 192);      // [4]
  float*          w2ff  = (float*)(ws + 1024);            // 262144 B: f32 B-frags of w2
  unsigned short* wp1f  = (unsigned short*)(ws + 263168); // 131072 B: bf16 B-frags of wp1
  float*          s_buf = (float*)(ws + 394240);          // 262144 B
  float*          score = (float*)(ws + 656384);          // 262144 B
  int*            topk  = (int*)(ws + 918528);            //  65536 B
  unsigned short* pf_bf = (unsigned short*)(ws + 984064); // 33554432 B: pf bf16 [N,256]

  k_zero    <<<1,   64,  0, stream>>>(g_acc);
  k_prep_w2 <<<128, 256, 0, stream>>>(w2, w2ff);
  k_prep_wp1<<<16,  256, 0, stream>>>(wp1, wp1f);
  k_mlp     <<<512, 256, 0, stream>>>(x, w1, b1, b2, wsmp, bsmp, w2ff, pf_bf, s_buf, g_acc);
  k_bnstats <<<1,   1,   0, stream>>>(g_acc, g_musd);
  k_score   <<<256, 256, 0, stream>>>(s_buf, g_musd, gamma, beta, score);
  k_sort    <<<4, 1024, PPB * 8 + 1024 * 4, stream>>>(score, topk, g_bmax, g_bsum);
  k_proc    <<<128, 256, 0, stream>>>(pf_bf, topk, score, g_bmax, g_bsum, wp1f, bp1, wp2, bp2, out);
}
